// Glm4MoeLiteDecoderLayer_83056077570287
// MI455X (gfx1250) — matrix operand order repaired and checked by reading, not compiled, not run on hardware
//
#include <hip/hip_runtime.h>
#include <hip/hip_bf16.h>
#include <math.h>

// ---- model dims ----
#define H_HEADS 20
#define DNOPE 192
#define DROPE 64
#define DQK 256
#define DV 256
#define HID 2048
#define QLORA 768
#define KVLORA 512
#define NEXP 64
#define TOPKE 4
#define FMOE 1536
#define FSH 1536
#define SEQ 512
#define KVD (DNOPE + DV)          // 448
#define KVN (H_HEADS * KVD)       // 8960
#define QN  (H_HEADS * DQK)       // 5120
#define EPS_F 1e-5f
#define RSF_F 1.8f
#define SCALE_F 0.0625f           // 256^-0.5

// ---- WMMA tile config ----
#define BM 128
#define BN 64
#define BK 32
#define BKP 40                    // padded LDS row (bank-conflict-free b128 frag loads)

typedef __attribute__((ext_vector_type(8)))  __bf16 v8bf;
typedef __attribute__((ext_vector_type(16))) __bf16 v16bf;
typedef __attribute__((ext_vector_type(8)))  float  v8f;

// pack two floats -> two bf16 (round-to-nearest, ties away) in 3 VALU ops:
// 2x v_add_u32 + 1x v_perm_b32 (result = bf16(y)<<16 | bf16(x))
__device__ __forceinline__ unsigned int pack2_bf16(float x, float y) {
    unsigned int ux = __builtin_bit_cast(unsigned int, x) + 0x8000u;
    unsigned int uy = __builtin_bit_cast(unsigned int, y) + 0x8000u;
    return __builtin_amdgcn_perm(uy, ux, 0x07060302u);
}
// float4 -> 4 packed bf16, one ds_store_b64
__device__ __forceinline__ void cvt_store8(unsigned short* dst, float4 v) {
    uint2 pk;
    pk.x = pack2_bf16(v.x, v.y);
    pk.y = pack2_bf16(v.z, v.w);
    *(uint2*)dst = pk;
}

// A fragment (16x32 bf16): lane m = lane&15; k-half base = (lane>>4)*8; chunks at kb, kb+16
__device__ __forceinline__ v16bf load_frag_a(const unsigned short* rowp, int lh) {
    const int kb = lh * 8;
    v8bf lo = *(const v8bf*)(rowp + kb);
    v8bf hi = *(const v8bf*)(rowp + kb + 16);
    return __builtin_shufflevector(lo, hi, 0,1,2,3,4,5,6,7,8,9,10,11,12,13,14,15);
}
// B fragment (32x16 bf16): lane n = lane&15; k-half base = (lane>>4)*8; chunks at kb, kb+16
__device__ __forceinline__ v16bf load_frag_b(const unsigned short* colp, int lh) {
    const int kb = lh * 8;
    v8bf lo = *(const v8bf*)(colp + kb);
    v8bf hi = *(const v8bf*)(colp + kb + 16);
    return __builtin_shufflevector(lo, hi, 0,1,2,3,4,5,6,7,8,9,10,11,12,13,14,15);
}

// ===== generic batched WMMA GEMM: C = scale*(A@B) (+add) =====
// REQUIRES: M % BM == 0, N % BN == 0, K % BK == 0 (true for every call site).
// Double-buffered LDS, register-staged global->LDS pipeline, 1 barrier per K-tile.
template<bool OPB_T, bool CAUSAL_SKIP>
__global__ __launch_bounds__(256) void gemm_wmma(
    const float* __restrict__ A, int lda, long long strideA,
    const float* __restrict__ B, int ldb, long long strideB,
    float* __restrict__ C, int ldc, long long strideC,
    const float* __restrict__ add, int ldadd,
    int M, int N, int K, float scale)
{
    const int n0 = blockIdx.x * BN;
    const int m0 = blockIdx.y * BM;
    if (CAUSAL_SKIP && n0 > m0 + BM - 1) return;   // fully-masked causal tile
    const long long z = blockIdx.z;
    A += z * strideA; B += z * strideB; C += z * strideC;

    __shared__ __align__(16) unsigned short Asb[2][BM * BKP];
    __shared__ __align__(16) unsigned short Bsb[2][BN * BKP];

    const int tid = threadIdx.x;
    const int lane = tid & 31, wave = tid >> 5;
    const int wm = wave & 3, wn = wave >> 2;
    const int ln = lane & 15, lh = lane >> 4;
    // B K-major staging micro-tile: 2(k) x 4(n) per thread
    const int bkk = (tid >> 4) << 1;    // 0,2,..,30
    const int bn4 = (tid & 15) << 2;    // 0,4,..,60

    v8f acc[2][2] = {};
    float4 ra[4], rb[2];

    auto gload = [&](int k0) {
#pragma unroll
        for (int j = 0; j < 4; ++j) {              // A: 128x32 fp32 = 1024 float4
            int i = tid + j * 256;
            int r = i >> 3, c4 = (i & 7) << 2;
            ra[j] = *(const float4*)&A[(long long)(m0 + r) * lda + k0 + c4];
        }
        if (OPB_T) {                               // B as N x K row-major
#pragma unroll
            for (int j = 0; j < 2; ++j) {
                int i = tid + j * 256;
                int n = i >> 3, c4 = (i & 7) << 2;
                rb[j] = *(const float4*)&B[(long long)(n0 + n) * ldb + k0 + c4];
            }
        } else {                                   // B as K x N row-major: 2 adjacent k rows
            rb[0] = *(const float4*)&B[(long long)(k0 + bkk) * ldb + n0 + bn4];
            rb[1] = *(const float4*)&B[(long long)(k0 + bkk + 1) * ldb + n0 + bn4];
        }
    };
    auto sstore = [&](int buf) {
#pragma unroll
        for (int j = 0; j < 4; ++j) {
            int i = tid + j * 256;
            int r = i >> 3, c4 = (i & 7) << 2;
            cvt_store8(&Asb[buf][r * BKP + c4], ra[j]);
        }
        if (OPB_T) {
#pragma unroll
            for (int j = 0; j < 2; ++j) {
                int i = tid + j * 256;
                int n = i >> 3, c4 = (i & 7) << 2;
                cvt_store8(&Bsb[buf][n * BKP + c4], rb[j]);
            }
        } else {                                   // transpose to [n][k]: pack k-pairs, b32 stores
            *(unsigned int*)&Bsb[buf][(bn4 + 0) * BKP + bkk] = pack2_bf16(rb[0].x, rb[1].x);
            *(unsigned int*)&Bsb[buf][(bn4 + 1) * BKP + bkk] = pack2_bf16(rb[0].y, rb[1].y);
            *(unsigned int*)&Bsb[buf][(bn4 + 2) * BKP + bkk] = pack2_bf16(rb[0].z, rb[1].z);
            *(unsigned int*)&Bsb[buf][(bn4 + 3) * BKP + bkk] = pack2_bf16(rb[0].w, rb[1].w);
        }
    };

    gload(0);
    int cur = 0;
    for (int k0 = 0; k0 < K; k0 += BK) {
        sstore(cur);
        __syncthreads();                           // single barrier per K-tile
        if (k0 + BK < K) gload(k0 + BK);           // overlap next-tile VMEM with WMMA

        v16bf af[2], bfr[2];
#pragma unroll
        for (int f = 0; f < 2; ++f) {
            af[f]  = load_frag_a(&Asb[cur][(wm * 32 + f * 16 + ln) * BKP], lh);
            bfr[f] = load_frag_b(&Bsb[cur][(wn * 32 + f * 16 + ln) * BKP], lh);
        }
#pragma unroll
        for (int fm = 0; fm < 2; ++fm)
#pragma unroll
            for (int fn = 0; fn < 2; ++fn)
                acc[fm][fn] = __builtin_amdgcn_wmma_f32_16x16x32_bf16(
                    false, af[fm], false, bfr[fn], (short)0, acc[fm][fn], false, false);
        cur ^= 1;
    }

    const int rb2 = lh * 8;
#pragma unroll
    for (int fm = 0; fm < 2; ++fm)
#pragma unroll
        for (int fn = 0; fn < 2; ++fn)
#pragma unroll
            for (int r = 0; r < 8; ++r) {
                int m = m0 + wm * 32 + fm * 16 + rb2 + r;
                int n = n0 + wn * 32 + fn * 16 + ln;
                float v = acc[fm][fn][r] * scale;
                if (add) v += add[(long long)m * ldadd + n];
                C[(long long)m * ldc + n] = v;
            }
}

// ===== gated up-projection: a = silu(X@Wg) * (X@Wu), optional token gather =====
// REQUIRES: N % BN == 0, K % BK == 0. M arbitrary (per-expert counts).
// A rows are CLAMPED, not guarded: garbage rows only feed C rows masked in the epilogue.
template<bool GATHER>
__global__ __launch_bounds__(256) void gated_up_wmma(
    const float* __restrict__ X, int ldx,
    const float* __restrict__ Wg, const float* __restrict__ Wu, int ldw, long long wstride,
    float* __restrict__ outA, int ldo, int N, int K, int Mfixed,
    const int* __restrict__ counts, const int* __restrict__ offsets, const int* __restrict__ perm)
{
    const int e = blockIdx.z;
    const int M = GATHER ? counts[e] : Mfixed;
    const int m0 = blockIdx.y * BM;
    if (m0 >= M) return;                            // empty expert M-tile -> early exit
    const int n0 = blockIdx.x * BN;
    const int slot0 = GATHER ? offsets[e] : 0;
    const float* wg = Wg + (long long)e * wstride;
    const float* wu = Wu + (long long)e * wstride;

    __shared__ __align__(16) unsigned short Asb[2][BM * BKP];
    __shared__ __align__(16) unsigned short Bgb[2][BN * BKP];
    __shared__ __align__(16) unsigned short Bub[2][BN * BKP];

    const int tid = threadIdx.x;
    const int lane = tid & 31, wave = tid >> 5;
    const int wm = wave & 3, wn = wave >> 2;
    const int ln = lane & 15, lh = lane >> 4;
    const int bkk = (tid >> 4) << 1;
    const int bn4 = (tid & 15) << 2;

    // K-invariant gathered row pointers (clamped to valid range)
    const float* rowp[4];
#pragma unroll
    for (int j = 0; j < 4; ++j) {
        int i = tid + j * 256;
        int r = i >> 3;
        int gm = m0 + r; if (gm > M - 1) gm = M - 1;
        int row = GATHER ? perm[slot0 + gm] : gm;
        rowp[j] = X + (long long)row * ldx + ((i & 7) << 2);
    }

    v8f accg[2][2] = {}, accu[2][2] = {};
    float4 ra[4], rg[2], ru[2];

    auto gload = [&](int k0) {
#pragma unroll
        for (int j = 0; j < 4; ++j) ra[j] = *(const float4*)(rowp[j] + k0);
        long long o0 = (long long)(k0 + bkk) * ldw + n0 + bn4;
        rg[0] = *(const float4*)&wg[o0];
        rg[1] = *(const float4*)&wg[o0 + ldw];
        ru[0] = *(const float4*)&wu[o0];
        ru[1] = *(const float4*)&wu[o0 + ldw];
    };
    auto sstore = [&](int buf) {
#pragma unroll
        for (int j = 0; j < 4; ++j) {
            int i = tid + j * 256;
            int r = i >> 3, c4 = (i & 7) << 2;
            cvt_store8(&Asb[buf][r * BKP + c4], ra[j]);
        }
        *(unsigned int*)&Bgb[buf][(bn4 + 0) * BKP + bkk] = pack2_bf16(rg[0].x, rg[1].x);
        *(unsigned int*)&Bgb[buf][(bn4 + 1) * BKP + bkk] = pack2_bf16(rg[0].y, rg[1].y);
        *(unsigned int*)&Bgb[buf][(bn4 + 2) * BKP + bkk] = pack2_bf16(rg[0].z, rg[1].z);
        *(unsigned int*)&Bgb[buf][(bn4 + 3) * BKP + bkk] = pack2_bf16(rg[0].w, rg[1].w);
        *(unsigned int*)&Bub[buf][(bn4 + 0) * BKP + bkk] = pack2_bf16(ru[0].x, ru[1].x);
        *(unsigned int*)&Bub[buf][(bn4 + 1) * BKP + bkk] = pack2_bf16(ru[0].y, ru[1].y);
        *(unsigned int*)&Bub[buf][(bn4 + 2) * BKP + bkk] = pack2_bf16(ru[0].z, ru[1].z);
        *(unsigned int*)&Bub[buf][(bn4 + 3) * BKP + bkk] = pack2_bf16(ru[0].w, ru[1].w);
    };

    gload(0);
    int cur = 0;
    for (int k0 = 0; k0 < K; k0 += BK) {
        sstore(cur);
        __syncthreads();
        if (k0 + BK < K) {
            gload(k0 + BK);
            __builtin_prefetch(&wg[(long long)(k0 + 2 * BK) * ldw + n0], 0, 1);
        }

        v16bf af[2], bg[2], bu[2];
#pragma unroll
        for (int f = 0; f < 2; ++f) {
            af[f] = load_frag_a(&Asb[cur][(wm * 32 + f * 16 + ln) * BKP], lh);
            bg[f] = load_frag_b(&Bgb[cur][(wn * 32 + f * 16 + ln) * BKP], lh);
            bu[f] = load_frag_b(&Bub[cur][(wn * 32 + f * 16 + ln) * BKP], lh);
        }
#pragma unroll
        for (int fm = 0; fm < 2; ++fm)
#pragma unroll
            for (int fn = 0; fn < 2; ++fn) {
                accg[fm][fn] = __builtin_amdgcn_wmma_f32_16x16x32_bf16(
                    false, af[fm], false, bg[fn], (short)0, accg[fm][fn], false, false);
                accu[fm][fn] = __builtin_amdgcn_wmma_f32_16x16x32_bf16(
                    false, af[fm], false, bu[fn], (short)0, accu[fm][fn], false, false);
            }
        cur ^= 1;
    }

    const int rb2 = lh * 8;
#pragma unroll
    for (int fm = 0; fm < 2; ++fm)
#pragma unroll
        for (int fn = 0; fn < 2; ++fn)
#pragma unroll
            for (int r = 0; r < 8; ++r) {
                int m = m0 + wm * 32 + fm * 16 + rb2 + r;
                int n = n0 + wn * 32 + fn * 16 + ln;
                if (m < M) {
                    float g = accg[fm][fn][r], u = accu[fm][fn][r];
                    float a = g / (1.f + __expf(-g)) * u;      // silu(g)*u
                    outA[(long long)(slot0 + m) * ldo + n] = a;
                }
            }
}

// ===== routed down-projection: out[token] += w_slot * (a_slots @ Wd_e) =====
__global__ __launch_bounds__(256) void moe_down_wmma(
    const float* __restrict__ Aslots, const float* __restrict__ Wd, long long wstride,
    float* __restrict__ out,
    const int* __restrict__ counts, const int* __restrict__ offsets,
    const int* __restrict__ perm, const float* __restrict__ slotw, int N, int K)
{
    const int e = blockIdx.z;
    const int M = counts[e];
    const int m0 = blockIdx.y * BM;
    if (m0 >= M) return;
    const int n0 = blockIdx.x * BN;
    const float* A = Aslots + (long long)offsets[e] * K;
    const float* B = Wd + (long long)e * wstride;

    __shared__ __align__(16) unsigned short Asb[2][BM * BKP];
    __shared__ __align__(16) unsigned short Bsb[2][BN * BKP];

    const int tid = threadIdx.x;
    const int lane = tid & 31, wave = tid >> 5;
    const int wm = wave & 3, wn = wave >> 2;
    const int ln = lane & 15, lh = lane >> 4;
    const int bkk = (tid >> 4) << 1;
    const int bn4 = (tid & 15) << 2;

    const float* rowp[4];                           // K-invariant, clamped rows
#pragma unroll
    for (int j = 0; j < 4; ++j) {
        int i = tid + j * 256;
        int r = i >> 3;
        int gm = m0 + r; if (gm > M - 1) gm = M - 1;
        rowp[j] = A + (long long)gm * K + ((i & 7) << 2);
    }

    v8f acc[2][2] = {};
    float4 ra[4], rb[2];

    auto gload = [&](int k0) {
#pragma unroll
        for (int j = 0; j < 4; ++j) ra[j] = *(const float4*)(rowp[j] + k0);
        long long o0 = (long long)(k0 + bkk) * N + n0 + bn4;
        rb[0] = *(const float4*)&B[o0];
        rb[1] = *(const float4*)&B[o0 + N];
    };
    auto sstore = [&](int buf) {
#pragma unroll
        for (int j = 0; j < 4; ++j) {
            int i = tid + j * 256;
            int r = i >> 3, c4 = (i & 7) << 2;
            cvt_store8(&Asb[buf][r * BKP + c4], ra[j]);
        }
        *(unsigned int*)&Bsb[buf][(bn4 + 0) * BKP + bkk] = pack2_bf16(rb[0].x, rb[1].x);
        *(unsigned int*)&Bsb[buf][(bn4 + 1) * BKP + bkk] = pack2_bf16(rb[0].y, rb[1].y);
        *(unsigned int*)&Bsb[buf][(bn4 + 2) * BKP + bkk] = pack2_bf16(rb[0].z, rb[1].z);
        *(unsigned int*)&Bsb[buf][(bn4 + 3) * BKP + bkk] = pack2_bf16(rb[0].w, rb[1].w);
    };

    gload(0);
    int cur = 0;
    for (int k0 = 0; k0 < K; k0 += BK) {
        sstore(cur);
        __syncthreads();
        if (k0 + BK < K) gload(k0 + BK);

        v16bf af[2], bfr[2];
#pragma unroll
        for (int f = 0; f < 2; ++f) {
            af[f]  = load_frag_a(&Asb[cur][(wm * 32 + f * 16 + ln) * BKP], lh);
            bfr[f] = load_frag_b(&Bsb[cur][(wn * 32 + f * 16 + ln) * BKP], lh);
        }
#pragma unroll
        for (int fm = 0; fm < 2; ++fm)
#pragma unroll
            for (int fn = 0; fn < 2; ++fn)
                acc[fm][fn] = __builtin_amdgcn_wmma_f32_16x16x32_bf16(
                    false, af[fm], false, bfr[fn], (short)0, acc[fm][fn], false, false);
        cur ^= 1;
    }

    const int rb2 = lh * 8;
#pragma unroll
    for (int fm = 0; fm < 2; ++fm)
#pragma unroll
        for (int fn = 0; fn < 2; ++fn)
#pragma unroll
            for (int r = 0; r < 8; ++r) {
                int m = m0 + wm * 32 + fm * 16 + rb2 + r;
                int n = n0 + wn * 32 + fn * 16 + ln;
                if (m < M) {
                    int slot = offsets[e] + m;
                    atomicAdd(&out[(long long)perm[slot] * HID + n], slotw[slot] * acc[fm][fn][r]);
                }
            }
}

// ================= small kernels =================
__global__ void rmsnorm_kernel(const float* __restrict__ x, const float* __restrict__ g,
                               float* __restrict__ y, int cols, int ldx, int ldy)
{
    const int row = blockIdx.x;
    const float* xr = x + (long long)row * ldx;
    float ss = 0.f;
    for (int c = threadIdx.x * 4; c < cols; c += 1024) {
        float4 v = *(const float4*)&xr[c];
        ss += v.x * v.x + v.y * v.y + v.z * v.z + v.w * v.w;
    }
    __shared__ float red[8];
    for (int o = 16; o > 0; o >>= 1) ss += __shfl_xor(ss, o, 32);
    if ((threadIdx.x & 31) == 0) red[threadIdx.x >> 5] = ss;
    __syncthreads();
    float tot = 0.f;
    for (int i = 0; i < 8; ++i) tot += red[i];
    float inv = rsqrtf(tot / (float)cols + EPS_F);
    float* yr = y + (long long)row * ldy;
    for (int c = threadIdx.x * 4; c < cols; c += 1024) {
        float4 v = *(const float4*)&xr[c];
        float4 gg = *(const float4*)&g[c];
        float4 o4;
        o4.x = v.x * inv * gg.x; o4.y = v.y * inv * gg.y;
        o4.z = v.z * inv * gg.z; o4.w = v.w * inv * gg.w;
        *(float4*)&yr[c] = o4;
    }
}

__global__ void rope_q_kernel(float* __restrict__ q)
{
    const int s = blockIdx.x, h = blockIdx.y, j = threadIdx.x; // 32 threads (one wave)
    float* x = q + (long long)s * QN + h * DQK + DNOPE;
    float a = x[2 * j], b = x[2 * j + 1];
    float inv = __expf(-(float)j * (9.210340372f / 32.f));     // 10000^(-j/32)
    float ang = (float)s * inv;
    float cs = __cosf(ang), sn = __sinf(ang);
    x[j]      = a * cs - b * sn;
    x[j + 32] = b * cs + a * sn;
}

__global__ void rope_k_kernel(const float* __restrict__ ckv, float* __restrict__ kpe)
{
    const int s = blockIdx.x, j = threadIdx.x;
    const float* x = ckv + (long long)s * (KVLORA + DROPE) + KVLORA;
    float a = x[2 * j], b = x[2 * j + 1];
    float inv = __expf(-(float)j * (9.210340372f / 32.f));
    float ang = (float)s * inv;
    float cs = __cosf(ang), sn = __sinf(ang);
    kpe[(long long)s * DROPE + j]      = a * cs - b * sn;
    kpe[(long long)s * DROPE + j + 32] = b * cs + a * sn;
}

__global__ void assemble_kf(const float* __restrict__ kv, const float* __restrict__ kpe,
                            float* __restrict__ kf)
{
    const int h = blockIdx.x, s = blockIdx.y;
    const int c4 = threadIdx.x * 4;                // 64 threads x float4 = 256 cols
    float4 v = (c4 < DNOPE) ? *(const float4*)&kv[(long long)s * KVN + h * KVD + c4]
                            : *(const float4*)&kpe[(long long)s * DROPE + (c4 - DNOPE)];
    *(float4*)&kf[((long long)h * SEQ + s) * DQK + c4] = v;
}

__global__ void softmax_causal(float* __restrict__ scores, int S)
{
    const int s = blockIdx.x, h = blockIdx.y;
    float* row = scores + ((long long)h * S + s) * (long long)S;
    const int n = s + 1;
    __shared__ float red[8];
    float mx = -3.4e38f;
    for (int c = threadIdx.x; c < n; c += 256) mx = fmaxf(mx, row[c]);
    for (int o = 16; o > 0; o >>= 1) mx = fmaxf(mx, __shfl_xor(mx, o, 32));
    if ((threadIdx.x & 31) == 0) red[threadIdx.x >> 5] = mx;
    __syncthreads();
    mx = red[0];
    for (int i = 1; i < 8; ++i) mx = fmaxf(mx, red[i]);
    float sum = 0.f;
    for (int c = threadIdx.x; c < n; c += 256) { float e = __expf(row[c] - mx); row[c] = e; sum += e; }
    for (int o = 16; o > 0; o >>= 1) sum += __shfl_xor(sum, o, 32);
    __syncthreads();
    if ((threadIdx.x & 31) == 0) red[threadIdx.x >> 5] = sum;
    __syncthreads();
    sum = 0.f;
    for (int i = 0; i < 8; ++i) sum += red[i];
    float inv = 1.f / sum;
    for (int c = threadIdx.x; c < n; c += 256) row[c] *= inv;
    for (int c = n + threadIdx.x; c < S; c += 256) row[c] = 0.f;  // masked -> exact 0
}

__global__ void zero_ints(int* p, int n)
{ int i = blockIdx.x * blockDim.x + threadIdx.x; if (i < n) p[i] = 0; }

__global__ void router_topk(const float* __restrict__ lg, const float* __restrict__ bias,
                            int* __restrict__ idx, float* __restrict__ wtok, int* __restrict__ counts)
{
    const int t = blockIdx.x * blockDim.x + threadIdx.x;
    if (t >= SEQ) return;
    float scv[NEXP], bv[NEXP];
    const float* lr = lg + (long long)t * NEXP;
    for (int e = 0; e < NEXP; ++e) {
        float v = 1.f / (1.f + __expf(-lr[e]));
        scv[e] = v; bv[e] = v + bias[e];
    }
    float ws = 0.f; float wk[TOPKE]; int sel[TOPKE];
    for (int k = 0; k < TOPKE; ++k) {
        float best = -3.4e38f; int bi = 0;
        for (int e = 0; e < NEXP; ++e) if (bv[e] > best) { best = bv[e]; bi = e; }
        sel[k] = bi; wk[k] = scv[bi]; ws += scv[bi]; bv[bi] = -3.4e38f;
    }
    float inv = RSF_F / (ws + 1e-20f);
    for (int k = 0; k < TOPKE; ++k) {
        idx[t * TOPKE + k] = sel[k];
        wtok[t * TOPKE + k] = wk[k] * inv;
        atomicAdd(&counts[sel[k]], 1);
    }
}

__global__ void prefix_scan(const int* __restrict__ counts, int* __restrict__ offsets)
{
    if (threadIdx.x == 0) { int o = 0; for (int e = 0; e < NEXP; ++e) { offsets[e] = o; o += counts[e]; } }
}

__global__ void scatter_tokens(const int* __restrict__ idx, const float* __restrict__ wtok,
                               const int* __restrict__ offsets, int* __restrict__ cursors,
                               int* __restrict__ perm, float* __restrict__ slw)
{
    const int t = blockIdx.x * blockDim.x + threadIdx.x;
    if (t >= SEQ) return;
    for (int k = 0; k < TOPKE; ++k) {
        int e = idx[t * TOPKE + k];
        int slot = offsets[e] + atomicAdd(&cursors[e], 1);
        perm[slot] = t; slw[slot] = wtok[t * TOPKE + k];
    }
}

// ================= host orchestration =================
extern "C" void kernel_launch(void* const* d_in, const int* in_sizes, int n_in,
                              void* d_out, int out_size, void* d_ws, size_t ws_size,
                              hipStream_t stream)
{
    (void)in_sizes; (void)n_in; (void)out_size; (void)ws_size;
    const float* hidden = (const float*)d_in[0];
    const float* g_in   = (const float*)d_in[1];
    const float* W_qa   = (const float*)d_in[2];
    const float* g_qa   = (const float*)d_in[3];
    const float* W_qb   = (const float*)d_in[4];
    const float* W_kva  = (const float*)d_in[5];
    const float* g_kva  = (const float*)d_in[6];
    const float* W_kvb  = (const float*)d_in[7];
    const float* W_o    = (const float*)d_in[8];
    const float* g_post = (const float*)d_in[9];
    const float* Wr     = (const float*)d_in[10];
    const float* bias_e = (const float*)d_in[11];
    const float* Wg_e   = (const float*)d_in[12];
    const float* Wu_e   = (const float*)d_in[13];
    const float* Wd_e   = (const float*)d_in[14];
    const float* Wg_s   = (const float*)d_in[15];
    const float* Wu_s   = (const float*)d_in[16];
    const float* Wd_s   = (const float*)d_in[17];
    float* out = (float*)d_out;

    float* p = (float*)d_ws;
    float* h    = p; p += (size_t)SEQ * HID;
    float* qa   = p; p += (size_t)SEQ * QLORA;
    float* q    = p; p += (size_t)SEQ * QN;
    float* ckv  = p; p += (size_t)SEQ * (KVLORA + DROPE);
    float* cn   = p; p += (size_t)SEQ * KVLORA;
    float* kv   = p; p += (size_t)SEQ * KVN;
    float* kpe  = p; p += (size_t)SEQ * DROPE;
    float* kf   = p; p += (size_t)H_HEADS * SEQ * DQK;
    float* sc   = p; p += (size_t)H_HEADS * SEQ * SEQ;
    float* ot   = p; p += (size_t)SEQ * QN;
    float* x    = p; p += (size_t)SEQ * HID;
    float* h2   = p; p += (size_t)SEQ * HID;
    float* lg   = p; p += (size_t)SEQ * NEXP;
    float* ash  = p; p += (size_t)SEQ * FSH;
    float* aslt = p; p += (size_t)SEQ * TOPKE * FMOE;
    float* slw  = p; p += (size_t)SEQ * TOPKE;
    float* wtok = p; p += (size_t)SEQ * TOPKE;
    int* idx     = (int*)p; p += SEQ * TOPKE;
    int* counts  = (int*)p; p += NEXP;
    int* cursors = (int*)p; p += NEXP;
    int* offsets = (int*)p; p += NEXP;
    int* perm    = (int*)p; p += SEQ * TOPKE;

    const dim3 blk(256);

    // 1. input rmsnorm
    rmsnorm_kernel<<<SEQ, blk, 0, stream>>>(hidden, g_in, h, HID, HID, HID);
    // 2. q LoRA down, norm, up
    gemm_wmma<false,false><<<dim3(QLORA/BN, SEQ/BM, 1), blk, 0, stream>>>(
        h, HID, 0, W_qa, QLORA, 0, qa, QLORA, 0, nullptr, 0, SEQ, QLORA, HID, 1.f);
    rmsnorm_kernel<<<SEQ, blk, 0, stream>>>(qa, g_qa, qa, QLORA, QLORA, QLORA);
    gemm_wmma<false,false><<<dim3(QN/BN, SEQ/BM, 1), blk, 0, stream>>>(
        qa, QLORA, 0, W_qb, QN, 0, q, QN, 0, nullptr, 0, SEQ, QN, QLORA, 1.f);
    // 3. kv LoRA down (+k_pe), norm, up
    gemm_wmma<false,false><<<dim3((KVLORA+DROPE)/BN, SEQ/BM, 1), blk, 0, stream>>>(
        h, HID, 0, W_kva, KVLORA+DROPE, 0, ckv, KVLORA+DROPE, 0, nullptr, 0,
        SEQ, KVLORA+DROPE, HID, 1.f);
    rmsnorm_kernel<<<SEQ, blk, 0, stream>>>(ckv, g_kva, cn, KVLORA, KVLORA+DROPE, KVLORA);
    gemm_wmma<false,false><<<dim3(KVN/BN, SEQ/BM, 1), blk, 0, stream>>>(
        cn, KVLORA, 0, W_kvb, KVN, 0, kv, KVN, 0, nullptr, 0, SEQ, KVN, KVLORA, 1.f);
    // 4. RoPE (q_pe in place, k_pe to side buffer) + assemble kf = [k_nope | k_pe]
    rope_q_kernel<<<dim3(SEQ, H_HEADS), 32, 0, stream>>>(q);
    rope_k_kernel<<<SEQ, 32, 0, stream>>>(ckv, kpe);
    assemble_kf<<<dim3(H_HEADS, SEQ), 64, 0, stream>>>(kv, kpe, kf);
    // 5. attention: scores = scale*qf@kf^T (skip strictly-upper tiles), softmax, o = P@V
    gemm_wmma<true,true><<<dim3(SEQ/BN, SEQ/BM, H_HEADS), blk, 0, stream>>>(
        q, QN, 256LL, kf, DQK, (long long)SEQ*DQK, sc, SEQ, (long long)SEQ*SEQ,
        nullptr, 0, SEQ, SEQ, DQK, SCALE_F);
    softmax_causal<<<dim3(SEQ, H_HEADS), 256, 0, stream>>>(sc, SEQ);
    gemm_wmma<false,false><<<dim3(DV/BN, SEQ/BM, H_HEADS), blk, 0, stream>>>(
        sc, SEQ, (long long)SEQ*SEQ, kv + DNOPE, KVN, (long long)KVD,
        ot, QN, (long long)DV, nullptr, 0, SEQ, DV, SEQ, 1.f);
    // 6. output proj + residual
    gemm_wmma<false,false><<<dim3(HID/BN, SEQ/BM, 1), blk, 0, stream>>>(
        ot, QN, 0, W_o, HID, 0, x, HID, 0, hidden, HID, SEQ, HID, QN, 1.f);
    // 7. post-attn norm, router, top-4 gather
    rmsnorm_kernel<<<SEQ, blk, 0, stream>>>(x, g_post, h2, HID, HID, HID);
    gemm_wmma<false,false><<<dim3(NEXP/BN, SEQ/BM, 1), blk, 0, stream>>>(
        h2, HID, 0, Wr, NEXP, 0, lg, NEXP, 0, nullptr, 0, SEQ, NEXP, HID, 1.f);
    zero_ints<<<1, 128, 0, stream>>>(counts, 2 * NEXP);      // counts + cursors (adjacent)
    router_topk<<<2, 256, 0, stream>>>(lg, bias_e, idx, wtok, counts);
    prefix_scan<<<1, 32, 0, stream>>>(counts, offsets);
    scatter_tokens<<<2, 256, 0, stream>>>(idx, wtok, offsets, cursors, perm, slw);
    // 8. shared expert: gated up, then down (+residual2) -> out
    gated_up_wmma<false><<<dim3(FSH/BN, SEQ/BM, 1), blk, 0, stream>>>(
        h2, HID, Wg_s, Wu_s, FSH, 0LL, ash, FSH, FSH, HID, SEQ, nullptr, nullptr, nullptr);
    gemm_wmma<false,false><<<dim3(HID/BN, SEQ/BM, 1), blk, 0, stream>>>(
        ash, FSH, 0, Wd_s, HID, 0, out, HID, 0, x, HID, SEQ, HID, FSH, 1.f);
    // 9. routed experts: grouped gated up (gather), weighted atomic down into out
    gated_up_wmma<true><<<dim3(FMOE/BN, SEQ/BM, NEXP), blk, 0, stream>>>(
        h2, HID, Wg_e, Wu_e, FMOE, (long long)HID * FMOE, aslt, FMOE, FMOE, HID, 0,
        counts, offsets, perm);
    moe_down_wmma<<<dim3(HID/BN, SEQ/BM, NEXP), blk, 0, stream>>>(
        aslt, Wd_e, (long long)FMOE * HID, out, counts, offsets, perm, slw, HID, FMOE);
}
